// Attention_69140383531299
// MI455X (gfx1250) — compile-verified, matrix operand order checked by reading, not run on hardware
//
#include <hip/hip_runtime.h>
#include <hip/hip_bf16.h>
#include <math.h>

// ---- problem constants (match reference) ----
#define HID  300
#define IMG  2048
#define NH   6
#define HD   50
#define NLAB 7
#define BSZ  16384
#define NPAD 320   // HID padded to multiple of 64 (N tiles) / 32 (K of fc GEMM)

typedef __attribute__((ext_vector_type(16))) __bf16 v16bf;
typedef __attribute__((ext_vector_type(8)))  __bf16 v8bf;
typedef __attribute__((ext_vector_type(8)))  float  v8f;

// ------------------------------------------------------------------
// pack 16 fp32 -> v16bf (compiler emits v_cvt_pk_bf16_f32 pairs)
// ------------------------------------------------------------------
__device__ inline v16bf cvt16(const float4& x0, const float4& x1,
                              const float4& x2, const float4& x3) {
    v16bf a;
    a[0]=(__bf16)x0.x;  a[1]=(__bf16)x0.y;  a[2]=(__bf16)x0.z;  a[3]=(__bf16)x0.w;
    a[4]=(__bf16)x1.x;  a[5]=(__bf16)x1.y;  a[6]=(__bf16)x1.z;  a[7]=(__bf16)x1.w;
    a[8]=(__bf16)x2.x;  a[9]=(__bf16)x2.y;  a[10]=(__bf16)x2.z; a[11]=(__bf16)x2.w;
    a[12]=(__bf16)x3.x; a[13]=(__bf16)x3.y; a[14]=(__bf16)x3.z; a[15]=(__bf16)x3.w;
    return a;
}

// Fragment loader: lane holds K = [off, off+8) and [off+16, off+24) of `row`
// (ISA 16-bit 16x32 A/B per-lane layout, off = k0 + half*8).
template <typename T>
__device__ inline v16bf load_frag(const T* __restrict__ row, int off) {
    if constexpr (sizeof(T) == 4) {
        const float* p = (const float*)row + off;
        float4 x0 = *(const float4*)(p);
        float4 x1 = *(const float4*)(p + 4);
        float4 x2 = *(const float4*)(p + 16);
        float4 x3 = *(const float4*)(p + 20);
        return cvt16(x0, x1, x2, x3);
    } else {
        const __bf16* p = (const __bf16*)row + off;
        v8bf lo = *(const v8bf*)(p);
        v8bf hi = *(const v8bf*)(p + 16);
        v16bf a;
        #pragma unroll
        for (int j = 0; j < 8; ++j) { a[j] = lo[j]; a[j + 8] = hi[j]; }
        return a;
    }
}

// ------------------------------------------------------------------
// Weight prep: Bt[NPAD][Kpad] (bf16) = transpose(Bm[Kb][N]) zero-padded.
// ------------------------------------------------------------------
__global__ void prep_bt(const float* __restrict__ Bm, int ldb, int Kb, int N,
                        int Kpad, __bf16* __restrict__ Bt) {
    int i = blockIdx.x * blockDim.x + threadIdx.x;
    if (i >= NPAD * Kpad) return;
    int n = i / Kpad, k = i % Kpad;
    float v = (n < N && k < Kb) ? Bm[(size_t)k * ldb + n] : 0.0f;
    Bt[i] = (__bf16)v;
}

// ------------------------------------------------------------------
// Tiny Q projection: [7x300] = [7x300] @ [300x300] + b
// ------------------------------------------------------------------
__global__ void qproj_kernel(const float* __restrict__ query,
                             const float* __restrict__ w,
                             const float* __restrict__ b,
                             float* __restrict__ Q) {
    int i = blockIdx.x * blockDim.x + threadIdx.x;
    if (i >= NLAB * HID) return;
    int l = i / HID, n = i % HID;
    float acc = b[n];
    for (int k = 0; k < HID; ++k)
        acc += query[l * HID + k] * w[k * HID + n];
    Q[i] = acc;
}

// ------------------------------------------------------------------
// Branch-free WMMA GEMM:  C[M x N] = A[M x K] @ Bt^T + bias
//  - A fp32 or bf16, fragments loaded straight from global (no LDS)
//  - Bt is pre-transposed zero-padded bf16 [NPAD][ldbt]
//  - block = 256 thr = 8 waves (4M x 2N); wave tile 64(M) x 32(N)
//    -> 8 v_wmma_f32_16x16x32_bf16 per wave per K-step
//  - requires M % 256 == 0, K % 32 == 0
// ------------------------------------------------------------------
template <typename TA>
__global__ __launch_bounds__(256) void gemm_bias_wmma(
    const TA* __restrict__ A, int lda, int K,
    const __bf16* __restrict__ Bt, int ldbt,
    const float* __restrict__ bias,
    float* __restrict__ C, int ldc, int N)
{
    const int t    = threadIdx.x;
    const int wave = t >> 5;
    const int lane = t & 31;
    const int half = lane >> 4;       // 0: lanes 0-15, 1: lanes 16-31
    const int lr   = lane & 15;
    const int wm   = wave >> 1;       // 0..3
    const int wn   = wave & 1;        // 0..1
    const int mbase = blockIdx.y * 256 + wm * 64;
    const int nbase = blockIdx.x * 64 + wn * 32;

    const TA* ap[4];
    const __bf16* bp[2];
    #pragma unroll
    for (int f = 0; f < 4; ++f) ap[f] = A + (size_t)(mbase + f * 16 + lr) * lda;
    #pragma unroll
    for (int g = 0; g < 2; ++g) bp[g] = Bt + (size_t)(nbase + g * 16 + lr) * ldbt;

    v8f acc[4][2] = {};

    for (int k0 = 0; k0 < K; k0 += 32) {
        const int off = k0 + half * 8;
        v16bf af[4], bfr[2];
        #pragma unroll
        for (int f = 0; f < 4; ++f) af[f] = load_frag<TA>(ap[f], off);
        #pragma unroll
        for (int g = 0; g < 2; ++g) bfr[g] = load_frag<__bf16>(bp[g], off);
        #pragma unroll
        for (int f = 0; f < 4; ++f)
            #pragma unroll
            for (int g = 0; g < 2; ++g)
                acc[f][g] = __builtin_amdgcn_wmma_f32_16x16x32_bf16(
                    false, af[f], false, bfr[g], (short)0, acc[f][g], false, false);
    }

    // epilogue: C/D layout VGPR r -> M = r + 8*half, N = lr (+tile)
    #pragma unroll
    for (int g = 0; g < 2; ++g) {
        int n = nbase + g * 16 + lr;
        if (n < N) {
            float bv = bias[n];
            #pragma unroll
            for (int f = 0; f < 4; ++f)
                #pragma unroll
                for (int r = 0; r < 8; ++r) {
                    int m = mbase + f * 16 + r + 8 * half;
                    C[(size_t)m * ldc + n] = acc[f][g][r] + bv;
                }
        }
    }
}

// ------------------------------------------------------------------
// Elementwise attention: per (b, label, head) softmax over head_dim=50
// x[b, lab, h*50+d] = softmax_d(Q*K/sqrt(50)) * V   -> bf16, K-padded row
// ------------------------------------------------------------------
__global__ void attn_kernel(const float* __restrict__ Q,
                            const float* __restrict__ Kb,
                            const float* __restrict__ Vb,
                            __bf16* __restrict__ xpad)
{
    int i = blockIdx.x * blockDim.x + threadIdx.x;
    if (i >= BSZ * NLAB * NH) return;
    int h   = i % NH;
    int lab = (i / NH) % NLAB;
    int b   = i / (NH * NLAB);

    const float inv_scale = 0.14142135623730950f; // 1/sqrt(50)
    const float* q = Q  + lab * HID + h * HD;
    const float* k = Kb + (size_t)b * HID + h * HD;
    const float* v = Vb + (size_t)b * HID + h * HD;

    float e[HD];
    float mx = -3.0e38f;
    #pragma unroll
    for (int d = 0; d < HD; ++d) {
        e[d] = q[d] * k[d] * inv_scale;
        mx = fmaxf(mx, e[d]);
    }
    float s = 0.0f;
    #pragma unroll
    for (int d = 0; d < HD; ++d) {
        e[d] = __expf(e[d] - mx);
        s += e[d];
    }
    float inv = 1.0f / s;
    __bf16* o = xpad + (size_t)(b * NLAB + lab) * NPAD + h * HD;
    #pragma unroll
    for (int d = 0; d < HD; ++d)
        o[d] = (__bf16)(e[d] * inv * v[d]);

    if (h == 0) { // zero the K-padding columns once per (b, lab) row
        __bf16* p = xpad + (size_t)(b * NLAB + lab) * NPAD + HID;
        #pragma unroll
        for (int j = 0; j < NPAD - HID; ++j) p[j] = (__bf16)0.0f;
    }
}

// ------------------------------------------------------------------
extern "C" void kernel_launch(void* const* d_in, const int* in_sizes, int n_in,
                              void* d_out, int out_size, void* d_ws, size_t ws_size,
                              hipStream_t stream) {
    const float* query = (const float*)d_in[0];
    const float* key   = (const float*)d_in[1];
    const float* value = (const float*)d_in[2];
    const float* wq_w  = (const float*)d_in[3];
    const float* wq_b  = (const float*)d_in[4];
    const float* wk_w  = (const float*)d_in[5];
    const float* wk_b  = (const float*)d_in[6];
    const float* wv_w  = (const float*)d_in[7];
    const float* wv_b  = (const float*)d_in[8];
    const float* fc_w  = (const float*)d_in[9];
    const float* fc_b  = (const float*)d_in[10];
    float* out = (float*)d_out;

    // workspace layout (all 16B-aligned, fully rewritten every call):
    char* ws = (char*)d_ws;
    const size_t kvBytes  = (size_t)BSZ * HID * sizeof(float);       // 19.66 MB
    const size_t xBytes   = (size_t)BSZ * NLAB * NPAD * 2;           // 73.40 MB
    const size_t btKBytes = (size_t)NPAD * IMG * 2;                  // 1.31 MB
    float*  Qbuf = (float*)(ws);                                     // 7*300 fp32
    float*  Kbuf = (float*)(ws + 16384);
    float*  Vbuf = (float*)(ws + 16384 + kvBytes);
    __bf16* xpad = (__bf16*)(ws + 16384 + 2 * kvBytes);
    __bf16* BtK  = (__bf16*)(ws + 16384 + 2 * kvBytes + xBytes);
    __bf16* BtV  = (__bf16*)(ws + 16384 + 2 * kvBytes + xBytes + btKBytes);
    __bf16* Btfc = (__bf16*)(ws + 16384 + 2 * kvBytes + xBytes + 2 * btKBytes);

    // 0) weight prep: transpose + pad to bf16 (tiny, L2-resident afterwards)
    prep_bt<<<(NPAD * IMG  + 255) / 256, 256, 0, stream>>>(wk_w, HID, IMG, HID, IMG,  BtK);
    prep_bt<<<(NPAD * IMG  + 255) / 256, 256, 0, stream>>>(wv_w, HID, IMG, HID, IMG,  BtV);
    prep_bt<<<(NPAD * NPAD + 255) / 256, 256, 0, stream>>>(fc_w, HID, HID, HID, NPAD, Btfc);

    // 1) Q projection (tiny)
    qproj_kernel<<<(NLAB * HID + 255) / 256, 256, 0, stream>>>(query, wq_w, wq_b, Qbuf);

    // 2) K and V projections: [16384 x 2048] @ [2048 x 300] + bias
    dim3 g1(NPAD / 64, BSZ / 256);   // (5, 64)
    gemm_bias_wmma<float><<<g1, 256, 0, stream>>>(key,   IMG, IMG, BtK, IMG, wk_b, Kbuf, HID, HID);
    gemm_bias_wmma<float><<<g1, 256, 0, stream>>>(value, IMG, IMG, BtV, IMG, wv_b, Vbuf, HID, HID);

    // 3) elementwise attention + softmax over head_dim, write bf16 padded x
    int nthr = BSZ * NLAB * NH;
    attn_kernel<<<(nthr + 255) / 256, 256, 0, stream>>>(Qbuf, Kbuf, Vbuf, xpad);

    // 4) output projection: [114688 x 320pad] @ [300 x 300] + bias
    dim3 g2(NPAD / 64, (BSZ * NLAB) / 256);   // (5, 448)
    gemm_bias_wmma<__bf16><<<g2, 256, 0, stream>>>(xpad, NPAD, NPAD, Btfc, NPAD, fc_b, out, HID, HID);
}
